// BondFastAttention_4432406250030
// MI455X (gfx1250) — matrix operand order repaired and checked by reading, not compiled, not run on hardware
//
#include <hip/hip_runtime.h>
#include <math.h>

// ---------------------------------------------------------------------------
// BondFastAttention for MI455X (gfx1250), wave32 + WMMA bf16.
//   B=64 graphs, L=1024 rows/graph, HID=512, HEADS=8, D=64.
// Pipeline: cvt X+weights -> QKV GEMMs (wmma bf16, 32x64 wave tiles)
//           -> global_q -> global_k -> kv/Wr/relu att (wmma)
//           -> Wo GEMM + bias + LayerNorm (wmma, fused epilogue).
// ---------------------------------------------------------------------------

#define HID   512
#define HEADS 8
#define DH    64
#define NB    64
#define LG    1024
#define MTOT  (NB * LG)          // 65536 rows
#define SCALE 0.125f             // 64^-0.5
#define EPS   1e-5f

typedef __attribute__((ext_vector_type(16))) __bf16          v16bf;
typedef __attribute__((ext_vector_type(8)))  float           v8f;
typedef __attribute__((ext_vector_type(8)))  unsigned short  ushort8;
typedef __attribute__((ext_vector_type(4)))  unsigned short  ushort4v;

// ---- bf16 <-> f32 helpers (round-to-nearest-even), storage = ushort -------
static __device__ __forceinline__ unsigned short f2bf_bits(float f) {
    unsigned u = __float_as_uint(f);
    unsigned r = u + 0x7FFFu + ((u >> 16) & 1u);
    return (unsigned short)(r >> 16);
}
static __device__ __forceinline__ float bf_bits2f(unsigned short h) {
    return __uint_as_float(((unsigned)h) << 16);
}
static __device__ __forceinline__ __bf16 bf_from_bits(unsigned short h) {
    __bf16 t; __builtin_memcpy(&t, &h, 2); return t;
}
static __device__ __forceinline__ __bf16 bf_from_f32(float f) {
    unsigned short h = f2bf_bits(f);
    return bf_from_bits(h);
}

// Build a v16bf fragment from two contiguous 8-element (16B) runs.
// Emits two global_load_b128 and bit-reinterprets into the WMMA operand.
static __device__ __forceinline__ v16bf load_frag16(const unsigned short* p0,
                                                    const unsigned short* p1) {
    ushort8 lo = *(const ushort8*)p0;
    ushort8 hi = *(const ushort8*)p1;
    v16bf out;
    __builtin_memcpy(&out, &lo, 16);
    __builtin_memcpy((char*)&out + 16, &hi, 16);
    return out;
}

// ISA 7.12.2: 16-bit A 16x32 — lane(0..15)=row M, lane half selects K groups:
//   element i<8  -> K = 8*half + i        (run at K = 8*half)
//   element i>=8 -> K = 16 + 8*half + (i-8)  (run at K = 16 + 8*half)
static __device__ __forceinline__ int a_kmap(int i, int half) {
    return (i < 8) ? (8 * half + i) : (8 + 8 * half + i);
}
// 16-bit B 32x16 — lane&15 = column N, element i -> K as in a_kmap, same as A (two 16B runs).
// C/D f32 16x16 — lane&15 = N, VGPR v -> M = v + 8*half.

// ---------------------------------------------------------------------------
// Kernel 1: f32 -> bf16 conversion, 4 elements/thread (b128 in, b64 out)
// ---------------------------------------------------------------------------
__global__ void cvt_f32_to_bf16_x4(const float* __restrict__ src,
                                   unsigned short* __restrict__ dst, int n) {
    int i = blockIdx.x * blockDim.x + threadIdx.x;
    int base = i * 4;
    if (base < n) {
        float4 f = ((const float4*)src)[i];
        ushort4v o;
        o.x = f2bf_bits(f.x);
        o.y = f2bf_bits(f.y);
        o.z = f2bf_bits(f.z);
        o.w = f2bf_bits(f.w);
        *(ushort4v*)(dst + base) = o;
    }
}

// ---------------------------------------------------------------------------
// Kernel 2: C_bf16[MTOT,512] = A_bf16[MTOT,512] @ W_bf16[512,512]^T
//   grid (MTOT/64, 512/256), block 256 = 8 waves arranged 2(M) x 4(N);
//   wave tile = 32x64 (2 M-tiles x 4 N-tiles, 8 wmma / K-step).
// ---------------------------------------------------------------------------
__global__ void gemm_bf16(const unsigned short* __restrict__ A,
                          const unsigned short* __restrict__ Wb,
                          unsigned short* __restrict__ Out) {
    const int tid  = threadIdx.x;
    const int wave = tid >> 5;
    const int lane = tid & 31;
    const int half = lane >> 4;
    const int r    = lane & 15;

    const int m0     = blockIdx.x * 64 + (wave & 1) * 32;
    const int n_base = blockIdx.y * 256 + (wave >> 1) * 64;

    v8f acc[2][4];
#pragma unroll
    for (int mm = 0; mm < 2; ++mm)
#pragma unroll
        for (int t = 0; t < 4; ++t) acc[mm][t] = (v8f)(0.0f);

    for (int k0 = 0; k0 < HID; k0 += 32) {
        v16bf a[2];
#pragma unroll
        for (int mm = 0; mm < 2; ++mm) {
            const unsigned short* arow =
                A + (size_t)(m0 + mm * 16 + r) * HID + k0;
            a[mm] = load_frag16(arow + 8 * half, arow + 16 + 8 * half);
        }
#pragma unroll
        for (int t = 0; t < 4; ++t) {
            const unsigned short* wrow =
                Wb + (size_t)(n_base + t * 16 + r) * HID + k0;
            v16bf bb = load_frag16(wrow + 8 * half, wrow + 16 + 8 * half);
#pragma unroll
            for (int mm = 0; mm < 2; ++mm)
                acc[mm][t] = __builtin_amdgcn_wmma_f32_16x16x32_bf16(
                    false, a[mm], false, bb, (short)0, acc[mm][t], false,
                    false);
        }
    }

#pragma unroll
    for (int mm = 0; mm < 2; ++mm)
#pragma unroll
        for (int t = 0; t < 4; ++t)
#pragma unroll
            for (int v = 0; v < 8; ++v) {
                int row = m0 + mm * 16 + v + 8 * half;
                int col = n_base + t * 16 + r;
                Out[(size_t)row * HID + col] = f2bf_bits(acc[mm][t][v]);
            }
}

// ---------------------------------------------------------------------------
// Kernels 3/4: softmax-over-D weighted reduction over L.
//   use_gq==0 : x = q[d]                (global_q pass, w = w_alpha)
//   use_gq==1 : x = gq[d] * k[d]        (global_k pass, w = w_beta)
//   gout[bh][d] = sum_l softmax_d(x*w*SCALE) * x
// ---------------------------------------------------------------------------
__global__ void global_reduce(const unsigned short* __restrict__ Xh,
                              const float* __restrict__ gq, int use_gq,
                              const float* __restrict__ wvec,
                              float* __restrict__ gout) {
    __shared__ float red[8][64];

    const int bh   = blockIdx.x;            // b*HEADS + h
    const int b    = bh / HEADS;
    const int h    = bh % HEADS;
    const int tid  = threadIdx.x;
    const int wave = tid >> 5;
    const int lane = tid & 31;

    const float w0 = wvec[lane] * SCALE;
    const float w1 = wvec[lane + 32] * SCALE;
    const float g0 = use_gq ? gq[bh * DH + lane] : 1.0f;
    const float g1 = use_gq ? gq[bh * DH + lane + 32] : 1.0f;

    float acc0 = 0.0f, acc1 = 0.0f;
    for (int l = wave; l < LG; l += 8) {
        const size_t base = (size_t)(b * LG + l) * HID + h * DH;
        float x0 = bf_bits2f(Xh[base + lane]) * g0;
        float x1 = bf_bits2f(Xh[base + lane + 32]) * g1;
        float t0 = x0 * w0;
        float t1 = x1 * w1;
        float m = fmaxf(t0, t1);
#pragma unroll
        for (int off = 16; off > 0; off >>= 1)
            m = fmaxf(m, __shfl_xor(m, off, 32));
        float e0 = __expf(t0 - m);
        float e1 = __expf(t1 - m);
        float s = e0 + e1;
#pragma unroll
        for (int off = 16; off > 0; off >>= 1)
            s += __shfl_xor(s, off, 32);
        float inv = 1.0f / s;
        acc0 += (e0 * inv) * x0;
        acc1 += (e1 * inv) * x1;
    }
    red[wave][lane]      = acc0;
    red[wave][lane + 32] = acc1;
    __syncthreads();
    if (tid < 64) {
        float s = 0.0f;
#pragma unroll
        for (int wv = 0; wv < 8; ++wv) s += red[wv][tid];
        gout[bh * DH + tid] = s;
    }
}

// ---------------------------------------------------------------------------
// Kernel 5: att = relu( (gk (.) v) @ Wr^T + q )  -> bf16
//   grid MTOT/16 blocks; wave = head; per wave 16x64 tile, K=64 (2 wmma steps)
// ---------------------------------------------------------------------------
__global__ void att_kernel(const unsigned short* __restrict__ Qbf,
                           const unsigned short* __restrict__ Vbf,
                           const float* __restrict__ gk,
                           const unsigned short* __restrict__ Wrb,
                           unsigned short* __restrict__ Attb) {
    const int tid  = threadIdx.x;
    const int h    = tid >> 5;              // wave index == head
    const int lane = tid & 31;
    const int half = lane >> 4;
    const int r    = lane & 15;

    const int m0 = blockIdx.x * 16;
    const int b  = m0 >> 10;                // tile never straddles a graph
    const float* gkh = gk + (size_t)(b * HEADS + h) * DH;

    v8f acc[4];
#pragma unroll
    for (int t = 0; t < 4; ++t) acc[t] = (v8f)(0.0f);

#pragma unroll
    for (int k0 = 0; k0 < DH; k0 += 32) {
        v16bf a;
        const unsigned short* vrow = Vbf + (size_t)(m0 + r) * HID + h * DH;
#pragma unroll
        for (int i = 0; i < 16; ++i) {
            int d = k0 + a_kmap(i, half);
            a[i] = bf_from_f32(gkh[d] * bf_bits2f(vrow[d]));
        }
#pragma unroll
        for (int t = 0; t < 4; ++t) {
            const unsigned short* wrow =
                Wrb + (size_t)(t * 16 + r) * DH + k0;  // Wr[e][d]
            v16bf bb = load_frag16(wrow + 8 * half, wrow + 16 + 8 * half);
            acc[t] = __builtin_amdgcn_wmma_f32_16x16x32_bf16(
                false, a, false, bb, (short)0, acc[t], false, false);
        }
    }

#pragma unroll
    for (int t = 0; t < 4; ++t) {
#pragma unroll
        for (int v = 0; v < 8; ++v) {
            int row = m0 + v + 8 * half;
            int col = h * DH + t * 16 + r;
            float q = bf_bits2f(Qbf[(size_t)row * HID + col]);
            float x = acc[t][v] + q;
            x = (x > 0.0f) ? x : 0.0f;
            Attb[(size_t)row * HID + col] = f2bf_bits(x);
        }
    }
}

// ---------------------------------------------------------------------------
// Kernel 6: out = LayerNorm( att @ Wo^T + bo ) * g + b   -> f32 d_out
//   block computes 16 full rows: 8 waves x 64 cols, stage in LDS, fuse LN.
// ---------------------------------------------------------------------------
__global__ void out_gemm_ln(const unsigned short* __restrict__ Attb,
                            const unsigned short* __restrict__ Wob,
                            const float* __restrict__ bo,
                            const float* __restrict__ ln_g,
                            const float* __restrict__ ln_b,
                            float* __restrict__ out) {
    __shared__ float smem[16][HID + 4];
    __shared__ float s_mu[16];
    __shared__ float s_rs[16];

    const int tid  = threadIdx.x;
    const int wave = tid >> 5;
    const int lane = tid & 31;
    const int half = lane >> 4;
    const int r    = lane & 15;

    const int m0     = blockIdx.x * 16;
    const int n_base = wave * 64;

    v8f acc[4];
#pragma unroll
    for (int t = 0; t < 4; ++t) acc[t] = (v8f)(0.0f);

    for (int k0 = 0; k0 < HID; k0 += 32) {
        const unsigned short* arow = Attb + (size_t)(m0 + r) * HID + k0;
        v16bf a = load_frag16(arow + 8 * half, arow + 16 + 8 * half);
#pragma unroll
        for (int t = 0; t < 4; ++t) {
            const unsigned short* wrow =
                Wob + (size_t)(n_base + t * 16 + r) * HID + k0;
            v16bf bb = load_frag16(wrow + 8 * half, wrow + 16 + 8 * half);
            acc[t] = __builtin_amdgcn_wmma_f32_16x16x32_bf16(
                false, a, false, bb, (short)0, acc[t], false, false);
        }
    }

#pragma unroll
    for (int t = 0; t < 4; ++t) {
#pragma unroll
        for (int v = 0; v < 8; ++v) {
            int rowl = v + 8 * half;
            int col  = n_base + t * 16 + r;
            smem[rowl][col] = acc[t][v] + bo[col];
        }
    }
    __syncthreads();

    if (tid < 16) {
        float s = 0.0f;
        for (int c = 0; c < HID; ++c) s += smem[tid][c];
        float mu = s * (1.0f / HID);
        float vs = 0.0f;
        for (int c = 0; c < HID; ++c) {
            float d = smem[tid][c] - mu;
            vs += d * d;
        }
        s_mu[tid] = mu;
        s_rs[tid] = rsqrtf(vs * (1.0f / HID) + EPS);
    }
    __syncthreads();

    for (int idx = tid; idx < 16 * HID; idx += 256) {
        int rowl = idx >> 9;
        int col  = idx & (HID - 1);
        float val = (smem[rowl][col] - s_mu[rowl]) * s_rs[rowl] * ln_g[col] +
                    ln_b[col];
        out[(size_t)(m0 + rowl) * HID + col] = val;
    }
}

// ---------------------------------------------------------------------------
// Host-side launch
// ---------------------------------------------------------------------------
extern "C" void kernel_launch(void* const* d_in, const int* in_sizes, int n_in,
                              void* d_out, int out_size, void* d_ws,
                              size_t ws_size, hipStream_t stream) {
    (void)in_sizes; (void)n_in; (void)out_size; (void)ws_size;

    const float* X       = (const float*)d_in[0];
    // d_in[1] = batch_scopes (int64) — contiguous equal-length, unused.
    const float* Wq      = (const float*)d_in[2];
    const float* Wk      = (const float*)d_in[3];
    const float* Wv      = (const float*)d_in[4];
    const float* Wr      = (const float*)d_in[5];
    const float* w_alpha = (const float*)d_in[6];
    const float* w_beta  = (const float*)d_in[7];
    const float* Wo      = (const float*)d_in[8];
    const float* bo      = (const float*)d_in[9];
    const float* ln_g    = (const float*)d_in[10];
    const float* ln_b    = (const float*)d_in[11];
    float* out           = (float*)d_out;

    // Workspace layout
    char* ws = (char*)d_ws;
    const size_t act_bytes = (size_t)MTOT * HID * 2;   // 64 MB each
    unsigned short* Xbf  = (unsigned short*)(ws);                    ws += act_bytes;
    unsigned short* Qbf  = (unsigned short*)(ws);                    ws += act_bytes;
    unsigned short* Kbf  = (unsigned short*)(ws);                    ws += act_bytes;
    unsigned short* Vbf  = (unsigned short*)(ws);                    ws += act_bytes;
    unsigned short* Attb = (unsigned short*)(ws);                    ws += act_bytes;
    unsigned short* Wqb  = (unsigned short*)(ws);                    ws += (size_t)HID * HID * 2;
    unsigned short* Wkb  = (unsigned short*)(ws);                    ws += (size_t)HID * HID * 2;
    unsigned short* Wvb  = (unsigned short*)(ws);                    ws += (size_t)HID * HID * 2;
    unsigned short* Wob  = (unsigned short*)(ws);                    ws += (size_t)HID * HID * 2;
    unsigned short* Wrb  = (unsigned short*)(ws);                    ws += (size_t)DH * DH * 2;
    float* gq            = (float*)(ws);                             ws += (size_t)NB * HEADS * DH * 4;
    float* gk            = (float*)(ws);

    // 1) activations + weights -> bf16 (X converted ONCE, reused by 3 GEMMs)
    const int nx = MTOT * HID;
    const int nw = HID * HID;
    cvt_f32_to_bf16_x4<<<(nx / 4 + 255) / 256, 256, 0, stream>>>(X, Xbf, nx);
    cvt_f32_to_bf16_x4<<<(nw / 4 + 255) / 256, 256, 0, stream>>>(Wq, Wqb, nw);
    cvt_f32_to_bf16_x4<<<(nw / 4 + 255) / 256, 256, 0, stream>>>(Wk, Wkb, nw);
    cvt_f32_to_bf16_x4<<<(nw / 4 + 255) / 256, 256, 0, stream>>>(Wv, Wvb, nw);
    cvt_f32_to_bf16_x4<<<(nw / 4 + 255) / 256, 256, 0, stream>>>(Wo, Wob, nw);
    cvt_f32_to_bf16_x4<<<(DH * DH / 4 + 255) / 256, 256, 0, stream>>>(Wr, Wrb,
                                                                      DH * DH);

    // 2) Q/K/V projections (WMMA bf16)
    dim3 ggrid(MTOT / 64, HID / 256);
    gemm_bf16<<<ggrid, 256, 0, stream>>>(Xbf, Wqb, Qbf);
    gemm_bf16<<<ggrid, 256, 0, stream>>>(Xbf, Wkb, Kbf);
    gemm_bf16<<<ggrid, 256, 0, stream>>>(Xbf, Wvb, Vbf);

    // 3) global_q , 4) global_k
    global_reduce<<<NB * HEADS, 256, 0, stream>>>(Qbf, gq, 0, w_alpha, gq);
    global_reduce<<<NB * HEADS, 256, 0, stream>>>(Kbf, gq, 1, w_beta, gk);

    // 5) att = relu((gk .* v) @ Wr^T + q)
    att_kernel<<<MTOT / 16, 256, 0, stream>>>(Qbf, Vbf, gk, Wrb, Attb);

    // 6) out GEMM + bias + LayerNorm
    out_gemm_ln<<<MTOT / 16, 256, 0, stream>>>(Attb, Wob, bo, ln_g, ln_b, out);
}
